// Transducer_89550068122301
// MI455X (gfx1250) — matrix operand order repaired and checked by reading, not compiled, not run on hardware
//
#include <hip/hip_runtime.h>
#include <hip/hip_bf16.h>
#include <math.h>

#define Bsz 4
#define Tt  300
#define Uu  64
#define U1  65
#define Hh  512
#define Vv  1024
#define NEGF (-1e30f)

typedef __attribute__((ext_vector_type(16))) __bf16 v16bf;
typedef __attribute__((ext_vector_type(8)))  float  v8f;

union Frag { uint4 q[2]; v16bf v; };
union Pk4  { __bf16 h[4]; uint2 u2; };

__device__ __forceinline__ float sigf(float x) { return 1.0f / (1.0f + __expf(-x)); }

__device__ __forceinline__ float laddexp(float x, float y) {
    float m = fmaxf(x, y);
    if (m < -5e29f) return NEGF;
    return m + __logf(__expf(x - m) + __expf(y - m));
}

// ---------------------------------------------------------------------------
// Generic GEMM: C[M,N] = A[M,K] * W[N,K]^T + bias, bf16 WMMA, f32 accumulate.
// Block = 256 threads (8 waves). WG tile 16(M) x 128(N), K step 32.
// Staging is float4-vectorized (all strides are 16B aligned, K % 4 == 0).
// ---------------------------------------------------------------------------
__global__ __launch_bounds__(256) void gemm_kernel(
    const float* __restrict__ A, int lda,
    const float* __restrict__ W, int ldw,
    const float* __restrict__ bias,
    float* __restrict__ C, int ldc,
    int M, int N, int K)
{
    __shared__ __align__(16) __bf16 As[16][32];
    __shared__ __align__(16) __bf16 Ws[128][32];
    const int tid  = threadIdx.x;
    const int wave = tid >> 5, lane = tid & 31;
    const int half = lane >> 4, mr = lane & 15;
    const int mbase = blockIdx.y * 16;
    const int nbase = blockIdx.x * 128;

    v8f acc = {};
    for (int kk = 0; kk < K; kk += 32) {
        // A tile: 16x32 = 128 float4s
        if (tid < 128) {
            int m = tid >> 3, k4 = (tid & 7) * 4;
            int gm = mbase + m, gk = kk + k4;
            float4 v = make_float4(0.f, 0.f, 0.f, 0.f);
            if (gm < M && gk < K) v = *(const float4*)&A[(size_t)gm * lda + gk];
            Pk4 p;
            p.h[0] = (__bf16)v.x; p.h[1] = (__bf16)v.y;
            p.h[2] = (__bf16)v.z; p.h[3] = (__bf16)v.w;
            *(uint2*)&As[m][k4] = p.u2;
        }
        // W tile: 128x32 = 1024 float4s, 4 per thread
        for (int i = tid; i < 1024; i += 256) {
            int n = i >> 3, k4 = (i & 7) * 4;
            int gn = nbase + n, gk = kk + k4;
            float4 v = make_float4(0.f, 0.f, 0.f, 0.f);
            if (gn < N && gk < K) v = *(const float4*)&W[(size_t)gn * ldw + gk];
            Pk4 p;
            p.h[0] = (__bf16)v.x; p.h[1] = (__bf16)v.y;
            p.h[2] = (__bf16)v.z; p.h[3] = (__bf16)v.w;
            *(uint2*)&Ws[n][k4] = p.u2;
        }
        __syncthreads();
        // A frag (ISA 16-bit A layout): lanes0-15 K{0-7,16-23}, lanes16-31 K{8-15,24-31}
        Frag af, bf;
        af.q[0] = *(const uint4*)&As[mr][half * 8];
        af.q[1] = *(const uint4*)&As[mr][16 + half * 8];
        // B frag: lanes0-15 hold K{0-7,16-23} of col n, lanes16-31 hold K{8-15,24-31}
        int nr = wave * 16 + mr;
        bf.q[0] = *(const uint4*)&Ws[nr][half * 8];
        bf.q[1] = *(const uint4*)&Ws[nr][16 + half * 8];
        acc = __builtin_amdgcn_wmma_f32_16x16x32_bf16(
                false, af.v, false, bf.v, (short)0, acc, false, false);
        __syncthreads();
    }
    int col = nbase + wave * 16 + mr;
    float bv = (bias != nullptr && col < N) ? bias[col] : 0.f;
#pragma unroll
    for (int r = 0; r < 8; ++r) {
        int row = mbase + half * 8 + r;           // C layout: VGPR r -> M=r / M=r+8
        if (row < M && col < N) C[(size_t)row * ldc + col] = acc[r] + bv;
    }
}

// ---------------------------------------------------------------------------
// Fused joint: A = tanh(E[b,t]+D[b,u]) (16 rows x K=512, bf16 in LDS);
// logits = A @ Wb^T + jb over V=1024; row-wise logsumexp; emit blank/label lp.
// Never materializes the (B,T,U1,V) logits tensor (saves ~640 MB of HBM traffic).
// ---------------------------------------------------------------------------
__global__ __launch_bounds__(256) void joint_kernel(
    const float* __restrict__ E, const float* __restrict__ D,
    const __bf16* __restrict__ Wb, const float* __restrict__ jb,
    const int* __restrict__ targets,
    float* __restrict__ blank_lp, float* __restrict__ lab_lp)
{
    __shared__ __align__(16) __bf16 Ash[16][Hh];
    __shared__ float redm[16][8];
    __shared__ float rowmaxS[16], blankS[16], labS[16];
    __shared__ int   tgtS[16];

    const int tid  = threadIdx.x;
    const int wave = tid >> 5, lane = tid & 31;
    const int half = lane >> 4, mr = lane & 15;
    const int utile = blockIdx.x, t = blockIdx.y, b = blockIdx.z;
    const int ubase = utile * 16;
    const int colbase = wave * 128;

    if (tid < 16) {
        int u = ubase + tid;
        tgtS[tid] = (u < Uu) ? targets[b * Uu + u] : 0;
        labS[tid] = 0.f; blankS[tid] = 0.f;
    }
    const float* Erow = E + (size_t)(b * Tt + t) * Hh;
    for (int i = tid; i < 16 * (Hh / 4); i += 256) {   // 2048 float4s
        int r = i >> 7, k4 = (i & 127) * 4;
        int u = ubase + r;
        Pk4 p;
        if (u < U1) {
            float4 e  = *(const float4*)&Erow[k4];
            float4 dd = *(const float4*)&D[(size_t)(b * U1 + u) * Hh + k4];
            p.h[0] = (__bf16)tanhf(e.x + dd.x);
            p.h[1] = (__bf16)tanhf(e.y + dd.y);
            p.h[2] = (__bf16)tanhf(e.z + dd.z);
            p.h[3] = (__bf16)tanhf(e.w + dd.w);
        } else {
            p.h[0] = p.h[1] = p.h[2] = p.h[3] = (__bf16)0.f;
        }
        *(uint2*)&Ash[r][k4] = p.u2;
    }
    __syncthreads();

    v8f acc[8] = {};
    const int koffA = half * 8;
    const int koffB = half * 8;
    for (int kk = 0; kk < Hh; kk += 32) {
        Frag af;
        af.q[0] = *(const uint4*)&Ash[mr][kk + koffA];
        af.q[1] = *(const uint4*)&Ash[mr][kk + 16 + koffA];
        if (kk + 32 < Hh)   // B panel is L2-resident; keep it warm in L0
            __builtin_prefetch(Wb + (size_t)(colbase + mr) * Hh + kk + 32, 0, 3);
#pragma unroll
        for (int s = 0; s < 8; ++s) {
            const __bf16* p = Wb + (size_t)(colbase + s * 16 + mr) * Hh + kk + koffB;
            Frag bf;
            bf.q[0] = *(const uint4*)p;
            bf.q[1] = *(const uint4*)(p + 16);
            acc[s] = __builtin_amdgcn_wmma_f32_16x16x32_bf16(
                        false, af.v, false, bf.v, (short)0, acc[s], false, false);
        }
    }
    // fold bias, extract blank/label entries, per-row max
    float rmax[8];
#pragma unroll
    for (int r = 0; r < 8; ++r) rmax[r] = -3.4e38f;
#pragma unroll
    for (int s = 0; s < 8; ++s) {
        int col = colbase + s * 16 + mr;
        float bv = jb[col];
#pragma unroll
        for (int r = 0; r < 8; ++r) {
            float v = acc[s][r] + bv;
            acc[s][r] = v;
            rmax[r] = fmaxf(rmax[r], v);
            int row = half * 8 + r;
            if (col == tgtS[row]) labS[row] = v;
        }
        if (col == 0) {
#pragma unroll
            for (int r = 0; r < 8; ++r) blankS[half * 8 + r] = acc[s][r];
        }
    }
#pragma unroll
    for (int r = 0; r < 8; ++r) {          // reduce across 16-lane half (wave32)
        float m = rmax[r];
        m = fmaxf(m, __shfl_xor(m, 1));
        m = fmaxf(m, __shfl_xor(m, 2));
        m = fmaxf(m, __shfl_xor(m, 4));
        m = fmaxf(m, __shfl_xor(m, 8));
        rmax[r] = m;
    }
    if (mr == 0) {
#pragma unroll
        for (int r = 0; r < 8; ++r) redm[half * 8 + r][wave] = rmax[r];
    }
    __syncthreads();
    if (tid < 16) {
        float m = redm[tid][0];
#pragma unroll
        for (int w = 1; w < 8; ++w) m = fmaxf(m, redm[tid][w]);
        rowmaxS[tid] = m;
    }
    __syncthreads();
    float rsum[8];
#pragma unroll
    for (int r = 0; r < 8; ++r) {
        float rm = rowmaxS[half * 8 + r];
        float s2 = 0.f;
#pragma unroll
        for (int s = 0; s < 8; ++s) s2 += __expf(acc[s][r] - rm);
        s2 += __shfl_xor(s2, 1);
        s2 += __shfl_xor(s2, 2);
        s2 += __shfl_xor(s2, 4);
        s2 += __shfl_xor(s2, 8);
        rsum[r] = s2;
    }
    __syncthreads();
    if (mr == 0) {
#pragma unroll
        for (int r = 0; r < 8; ++r) redm[half * 8 + r][wave] = rsum[r];
    }
    __syncthreads();
    if (tid < 16) {
        int u = ubase + tid;
        if (u < U1) {
            float s2 = 0.f;
            for (int w = 0; w < 8; ++w) s2 += redm[tid][w];
            float lse = rowmaxS[tid] + __logf(s2);
            size_t base = (size_t)(b * Tt + t);
            blank_lp[base * U1 + u] = blankS[tid] - lse;
            if (u < Uu) lab_lp[base * Uu + u] = labS[tid] - lse;
        }
    }
}

// ---------------------------------------------------------------------------
// Sequential LSTM over T steps, h/c in LDS. 1024 threads, 2 cells/thread.
// WT is gate-interleaved: WT[(k*512+u)*4 + g]  ->  one b128 load per 4-wide K.
// ---------------------------------------------------------------------------
__global__ __launch_bounds__(1024) void lstm_kernel(
    const float* __restrict__ xg, const float* __restrict__ WT,
    float* __restrict__ hs, int T_)
{
    __shared__ float hsm[4][Hh];
    __shared__ float csm[4][Hh];
    int tid = threadIdx.x;
    for (int i = tid; i < 4 * Hh; i += 1024) { hsm[i >> 9][i & 511] = 0.f; csm[i >> 9][i & 511] = 0.f; }
    __syncthreads();
    int u  = tid & 511;
    int b0 = tid >> 9;     // 0 or 1
    int b1 = b0 + 2;
    for (int t = 0; t < T_; ++t) {
        float a0[4], a1[4];
#pragma unroll
        for (int g = 0; g < 4; ++g) {
            a0[g] = xg[(size_t)(b0 * T_ + t) * 2048 + g * 512 + u];
            a1[g] = xg[(size_t)(b1 * T_ + t) * 2048 + g * 512 + u];
        }
        for (int k = 0; k < 512; k += 4) {
            float4 h0v = *(const float4*)&hsm[b0][k];
            float4 h1v = *(const float4*)&hsm[b1][k];
            float h0a[4] = {h0v.x, h0v.y, h0v.z, h0v.w};
            float h1a[4] = {h1v.x, h1v.y, h1v.z, h1v.w};
            const float4* wr = (const float4*)(WT + ((size_t)k * 512 + u) * 4);
#pragma unroll
            for (int j = 0; j < 4; ++j) {
                float4 w4 = wr[(size_t)j * 512];
                a0[0] = fmaf(h0a[j], w4.x, a0[0]);
                a0[1] = fmaf(h0a[j], w4.y, a0[1]);
                a0[2] = fmaf(h0a[j], w4.z, a0[2]);
                a0[3] = fmaf(h0a[j], w4.w, a0[3]);
                a1[0] = fmaf(h1a[j], w4.x, a1[0]);
                a1[1] = fmaf(h1a[j], w4.y, a1[1]);
                a1[2] = fmaf(h1a[j], w4.z, a1[2]);
                a1[3] = fmaf(h1a[j], w4.w, a1[3]);
            }
        }
        float c0 = sigf(a0[1]) * csm[b0][u] + sigf(a0[0]) * tanhf(a0[2]);
        float h0n = sigf(a0[3]) * tanhf(c0);
        float c1 = sigf(a1[1]) * csm[b1][u] + sigf(a1[0]) * tanhf(a1[2]);
        float h1n = sigf(a1[3]) * tanhf(c1);
        __syncthreads();
        hsm[b0][u] = h0n; csm[b0][u] = c0;
        hsm[b1][u] = h1n; csm[b1][u] = c1;
        hs[(size_t)(b0 * T_ + t) * 512 + u] = h0n;
        hs[(size_t)(b1 * T_ + t) * 512 + u] = h1n;
        __syncthreads();
    }
}

// ---------------------------------------------------------------------------
// Whh (2048 x 512, row j = g*512+u) -> gate-interleaved WT[(k*512+u)*4 + g].
__global__ void transpose_gates(const float* __restrict__ W, float* __restrict__ WT) {
    int idx = blockIdx.x * 256 + threadIdx.x;
    if (idx < 2048 * 512) {
        int j = idx >> 9, k = idx & 511;
        int g = j >> 9, uu = j & 511;
        WT[(((size_t)k * 512 + uu) << 2) + g] = W[idx];
    }
}

__global__ void cvt_bf16(const float* __restrict__ X, __bf16* __restrict__ Y, int n4) {
    int i = blockIdx.x * 256 + threadIdx.x;     // one float4 per thread
    if (i < n4) {
        float4 v = *(const float4*)&X[i * 4];
        Pk4 p;
        p.h[0] = (__bf16)v.x; p.h[1] = (__bf16)v.y;
        p.h[2] = (__bf16)v.z; p.h[3] = (__bf16)v.w;
        *(uint2*)&Y[i * 4] = p.u2;
    }
}

__global__ void reverse_pad(const float* __restrict__ x, float* __restrict__ y,
                            const int* __restrict__ len, int T_, int C_) {
    int t = blockIdx.x, b = blockIdx.y;
    int L = len[b];
    int src = (t < L) ? (L - 1 - t) : t;
    for (int c = threadIdx.x * 4; c < C_; c += blockDim.x * 4)
        *(float4*)&y[(size_t)(b * T_ + t) * C_ + c] =
            *(const float4*)&x[(size_t)(b * T_ + src) * C_ + c];
}

__global__ void concat_mask(const float* __restrict__ f, const float* __restrict__ bw,
                            float* __restrict__ y, const int* __restrict__ len) {
    int t = blockIdx.x, b = blockIdx.y;
    int c = threadIdx.x * 4;                     // 256 threads cover 1024
    float4 v = (c < 512) ? *(const float4*)&f[(size_t)(b * Tt + t) * 512 + c]
                         : *(const float4*)&bw[(size_t)(b * Tt + t) * 512 + (c - 512)];
    if (t >= len[b]) v = make_float4(0.f, 0.f, 0.f, 0.f);
    *(float4*)&y[(size_t)(b * Tt + t) * 1024 + c] = v;
}

__global__ void embed_kernel(const int* __restrict__ targets, const float* __restrict__ emb,
                             float* __restrict__ demb) {
    int u = blockIdx.x, b = blockIdx.y;
    int c = threadIdx.x * 4;                     // 128 threads cover 512
    int row = (u == 0) ? 0 : targets[b * Uu + (u - 1)];
    *(float4*)&demb[(size_t)(b * U1 + u) * Hh + c] =
        *(const float4*)&emb[(size_t)row * Hh + c];
}

__global__ void dmask_kernel(float* __restrict__ dh, const int* __restrict__ ulen) {
    int u = blockIdx.x, b = blockIdx.y;
    int c = threadIdx.x * 4;
    if (u >= ulen[b] + 1)
        *(float4*)&dh[(size_t)(b * U1 + u) * Hh + c] = make_float4(0.f, 0.f, 0.f, 0.f);
}

// ---------------------------------------------------------------------------
// RNNT alpha recursion + loss. One block of 512 threads (4 b-groups x 128).
// ---------------------------------------------------------------------------
__global__ __launch_bounds__(512) void rnnt_loss_kernel(
    const float* __restrict__ blank_lp, const float* __restrict__ lab_lp,
    const int* __restrict__ tlen, const int* __restrict__ ulen,
    float* __restrict__ out)
{
    __shared__ float alpha[4][U1];
    __shared__ float sbuf[4][U1];
    __shared__ float cc[4][U1];
    __shared__ float ll[4];
    int tid = threadIdx.x;
    int b = tid >> 7, u = tid & 127;
    bool act = (u < U1);
    if (act && u == 0) ll[b] = 0.f;
    __syncthreads();
    for (int t = 0; t < Tt; ++t) {
        float a = NEGF;
        if (act) {
            if (t == 0) a = (u == 0) ? 0.f : NEGF;
            else        a = alpha[b][u] + blank_lp[(size_t)(b * Tt + (t - 1)) * U1 + u];
        }
        float d = 0.f;
        if (act && u > 0) d = lab_lp[(size_t)(b * Tt + t) * Uu + (u - 1)];
        if (act) cc[b][u] = d;
        __syncthreads();
        for (int off = 1; off < U1; off <<= 1) {      // inclusive prefix-sum -> C
            float v = 0.f;
            if (act && u >= off) v = cc[b][u - off];
            __syncthreads();
            if (act) cc[b][u] += v;
            __syncthreads();
        }
        float Cv = act ? cc[b][u] : 0.f;
        if (act) sbuf[b][u] = a - Cv;
        __syncthreads();
        for (int off = 1; off < U1; off <<= 1) {      // inclusive cum-logsumexp
            float v = NEGF;
            if (act && u >= off) v = sbuf[b][u - off];
            __syncthreads();
            if (act) sbuf[b][u] = laddexp(sbuf[b][u], v);
            __syncthreads();
        }
        if (act) alpha[b][u] = Cv + sbuf[b][u];
        __syncthreads();
        if (act && u == 0 && t == tlen[b] - 1) {
            int uu = ulen[b];
            ll[b] = alpha[b][uu] + blank_lp[(size_t)(b * Tt + t) * U1 + uu];
        }
        __syncthreads();
    }
    if (tid == 0) out[0] = -(ll[0] + ll[1] + ll[2] + ll[3]) * 0.25f;
}

// ---------------------------------------------------------------------------
extern "C" void kernel_launch(void* const* d_in, const int* in_sizes, int n_in,
                              void* d_out, int out_size, void* d_ws, size_t ws_size,
                              hipStream_t stream) {
    (void)in_sizes; (void)n_in; (void)out_size; (void)ws_size;
    const float* inputs     = (const float*)d_in[0];
    const int*   ilen       = (const int*)  d_in[1];
    const int*   targets    = (const int*)  d_in[2];
    const int*   tglen      = (const int*)  d_in[3];
    const float* enc_Wih    = (const float*)d_in[4];
    const float* enc_Whh    = (const float*)d_in[5];
    const float* enc_bias   = (const float*)d_in[6];
    const float* enc_proj_W = (const float*)d_in[7];
    const float* enc_proj_b = (const float*)d_in[8];
    const float* dec_emb    = (const float*)d_in[9];
    const float* dec_Wih    = (const float*)d_in[10];
    const float* dec_Whh    = (const float*)d_in[11];
    const float* dec_bias   = (const float*)d_in[12];
    const float* dec_proj_W = (const float*)d_in[13];
    const float* dec_proj_b = (const float*)d_in[14];
    const float* j_fwd_W    = (const float*)d_in[15];
    const float* j_fwd_b    = (const float*)d_in[16];
    const float* j_proj_W   = (const float*)d_in[17];
    const float* j_proj_b   = (const float*)d_in[18];

    char* ws = (char*)d_ws;
    size_t off = 0;
    auto alloc = [&](size_t bytes) -> void* {
        void* p = ws + off;
        off += (bytes + 255) & ~(size_t)255;
        return p;
    };

    __bf16* Wb   = (__bf16*)alloc((size_t)Vv * Hh * 2);
    float* xA    = (float*)alloc((size_t)Bsz * Tt * 1024 * 4);
    float* xB    = (float*)alloc((size_t)Bsz * Tt * 1024 * 4);
    float* xrev  = (float*)alloc((size_t)Bsz * Tt * 1024 * 4);
    float* big0  = (float*)alloc((size_t)Bsz * Tt * 2048 * 4);   // xg_f / later enc,E
    float* big1  = (float*)alloc((size_t)Bsz * Tt * 2048 * 4);   // xg_b / later dec side
    float* hsf   = (float*)alloc((size_t)Bsz * Tt * 512 * 4);
    float* hsb   = (float*)alloc((size_t)Bsz * Tt * 512 * 4);
    float* WhhT  = (float*)alloc((size_t)512 * 2048 * 4);

    float* hsr   = xrev;                                 // free after xg_b gemm
    float* enc   = big0;                                 // B*T*512
    float* Emat  = big0 + (size_t)Bsz * Tt * 512;        // B*T*512
    float* demb  = big1;                                 // B*U1*512
    float* xg_d  = big1 + (size_t)Bsz * U1 * 512;        // B*U1*2048
    float* dh    = xg_d + (size_t)Bsz * U1 * 2048;       // B*U1*512
    float* decp  = dh   + (size_t)Bsz * U1 * 512;        // B*U1*512
    float* Dmat  = decp + (size_t)Bsz * U1 * 512;        // B*U1*512
    float* blank = xB;                                   // free after layer-2 concat
    float* lab   = xB + (size_t)Bsz * Tt * U1;

    const int M_enc = Bsz * Tt;       // 1200
    const int M_dec = Bsz * U1;       // 260

    // ---------------- Encoder: 3 BiLSTM layers ----------------
    const float* in = inputs;
    int Din = 80;
    float* outx = xA;
    for (int l = 0; l < 3; ++l) {
        const float* Wih_f = enc_Wih + (size_t)(l * 2 + 0) * 2048 * 1024;
        const float* Wih_b = enc_Wih + (size_t)(l * 2 + 1) * 2048 * 1024;
        const float* bfv   = enc_bias + (size_t)(l * 2 + 0) * 2048;
        const float* bbv   = enc_bias + (size_t)(l * 2 + 1) * 2048;

        gemm_kernel<<<dim3(2048 / 128, (M_enc + 15) / 16), 256, 0, stream>>>(
            in, Din, Wih_f, 1024, bfv, big0, 2048, M_enc, 2048, Din);
        reverse_pad<<<dim3(Tt, Bsz), 256, 0, stream>>>(in, xrev, ilen, Tt, Din);
        gemm_kernel<<<dim3(2048 / 128, (M_enc + 15) / 16), 256, 0, stream>>>(
            xrev, Din, Wih_b, 1024, bbv, big1, 2048, M_enc, 2048, Din);

        transpose_gates<<<(2048 * 512 + 255) / 256, 256, 0, stream>>>(
            enc_Whh + (size_t)(l * 2 + 0) * 2048 * 512, WhhT);
        lstm_kernel<<<1, 1024, 0, stream>>>(big0, WhhT, hsf, Tt);
        transpose_gates<<<(2048 * 512 + 255) / 256, 256, 0, stream>>>(
            enc_Whh + (size_t)(l * 2 + 1) * 2048 * 512, WhhT);
        lstm_kernel<<<1, 1024, 0, stream>>>(big1, WhhT, hsb, Tt);

        reverse_pad<<<dim3(Tt, Bsz), 256, 0, stream>>>(hsb, hsr, ilen, Tt, 512);
        concat_mask<<<dim3(Tt, Bsz), 256, 0, stream>>>(hsf, hsr, outx, ilen);
        in = outx; Din = 1024;
        outx = (outx == xA) ? xB : xA;
    }
    const float* xfin = in;   // xA

    // enc projection + pre-joint E (bias = j_fwd_b folded here)
    gemm_kernel<<<dim3(512 / 128, (M_enc + 15) / 16), 256, 0, stream>>>(
        xfin, 1024, enc_proj_W, 1024, enc_proj_b, enc, 512, M_enc, 512, 1024);
    gemm_kernel<<<dim3(512 / 128, (M_enc + 15) / 16), 256, 0, stream>>>(
        enc, 512, j_fwd_W, 1024, j_fwd_b, Emat, 512, M_enc, 512, 512);

    // ---------------- Decoder ----------------
    embed_kernel<<<dim3(U1, Bsz), 128, 0, stream>>>(targets, dec_emb, demb);
    gemm_kernel<<<dim3(2048 / 128, (M_dec + 15) / 16), 256, 0, stream>>>(
        demb, 512, dec_Wih, 512, dec_bias, xg_d, 2048, M_dec, 2048, 512);
    transpose_gates<<<(2048 * 512 + 255) / 256, 256, 0, stream>>>(dec_Whh, WhhT);
    lstm_kernel<<<1, 1024, 0, stream>>>(xg_d, WhhT, dh, U1);
    dmask_kernel<<<dim3(U1, Bsz), 128, 0, stream>>>(dh, tglen);
    gemm_kernel<<<dim3(512 / 128, (M_dec + 15) / 16), 256, 0, stream>>>(
        dh, 512, dec_proj_W, 512, dec_proj_b, decp, 512, M_dec, 512, 512);
    gemm_kernel<<<dim3(512 / 128, (M_dec + 15) / 16), 256, 0, stream>>>(
        decp, 512, j_fwd_W + 512, 1024, nullptr, Dmat, 512, M_dec, 512, 512);

    // ---------------- Fused joint + log-softmax gather ----------------
    cvt_bf16<<<(Vv * Hh / 4 + 255) / 256, 256, 0, stream>>>(j_proj_W, Wb, Vv * Hh / 4);
    joint_kernel<<<dim3((U1 + 15) / 16, Tt, Bsz), 256, 0, stream>>>(
        Emat, Dmat, Wb, j_proj_b, targets, blank, lab);

    // ---------------- RNNT loss ----------------
    rnnt_loss_kernel<<<1, 512, 0, stream>>>(blank, lab, ilen, tglen, (float*)d_out);
}
